// CNN_LSTM_4501125726526
// MI455X (gfx1250) — matrix operand order repaired and checked by reading, not compiled, not run on hardware
//
#include <hip/hip_runtime.h>
#include <math.h>

typedef unsigned short u16;
typedef unsigned int   u32;
typedef unsigned long long u64;
typedef __attribute__((ext_vector_type(16))) __bf16 v16bf;
typedef __attribute__((ext_vector_type(8)))  __bf16 v8bf;
typedef __attribute__((ext_vector_type(8)))  float  v8f;
typedef __attribute__((ext_vector_type(4)))  u32    u32x4;

#define BB 32     // batch
#define HH 168    // encoder timesteps
#define NN 512    // hidden
#define NF 4096   // N*F conv input
#define GG 2048   // 4*N gates
#define PP 24     // decoder steps
#define MROWS (BB*HH)   // 5376
#define NWG_LSTM 64     // persistent workgroups; each owns 8 hidden units

__device__ __forceinline__ u16 f2bf(float f) {
    u32 u = __float_as_uint(f);
    u32 r = 0x7FFFu + ((u >> 16) & 1u);
    return (u16)((u + r) >> 16);
}

__device__ __forceinline__ v8f wmma_bf16(v8bf a0, v8bf a1, v8bf b0, v8bf b1, v8f c) {
    union { v16bf v; v8bf h[2]; } A, B;
    A.h[0] = a0; A.h[1] = a1;
    B.h[0] = b0; B.h[1] = b1;
    return __builtin_amdgcn_wmma_f32_16x16x32_bf16(false, A.v, false, B.v,
                                                   (short)0, c, false, false);
}

// A operand supplied as two raw 128-bit loads (asm outputs), B as v8bf pair.
__device__ __forceinline__ v8f wmma_ab(u32x4 lo, u32x4 hi, v8bf b0, v8bf b1, v8f c) {
    union { u32x4 q[2]; v16bf v; } A;
    union { v8bf h[2];  v16bf v; } B;
    A.q[0] = lo; A.q[1] = hi;
    B.h[0] = b0; B.h[1] = b1;
    return __builtin_amdgcn_wmma_f32_16x16x32_bf16(false, A.v, false, B.v,
                                                   (short)0, c, false, false);
}

// ---------------------------------------------------------------- conversions
__global__ __launch_bounds__(256)
void cvt_bf16_k(const float* __restrict__ in, u16* __restrict__ out, int n) {
    for (int i = blockIdx.x * 256 + threadIdx.x; i < n; i += gridDim.x * 256)
        out[i] = f2bf(in[i]);
}

// in: [R][C] f32 row-major  ->  outT: [C][R] bf16
__global__ __launch_bounds__(256)
void transpose_bf16_k(const float* __restrict__ in, u16* __restrict__ outT,
                      int R, int C) {
    int idx = blockIdx.x * 256 + threadIdx.x;
    if (idx < R * C) {
        int r = idx / C, c = idx % C;
        outT[(size_t)c * R + r] = f2bf(in[idx]);
    }
}

__global__ __launch_bounds__(256)
void transpose_sum_bf16_k(const float* __restrict__ a, const float* __restrict__ b,
                          u16* __restrict__ outT, int R, int C) {
    int idx = blockIdx.x * 256 + threadIdx.x;
    if (idx < R * C) {
        int r = idx / C, c = idx % C;
        outT[(size_t)c * R + r] = f2bf(a[idx] + b[idx]);
    }
}

__global__ void init_bar_k(u32* bar) {
    if (threadIdx.x < 8) bar[threadIdx.x] = 0;
}

// ---------------------------------------------------------------- tiled GEMM
// Out[M][N] = A[M][K](bf16) * Bt[N][K](bf16)^T + bias.
// Block tile 64x128, 8 waves (wave = 16x64, 4 accumulators), K-chunk 32,
// double-buffered LDS with register prefetch of the next chunk.
template <int OUT_BF16>
__global__ __launch_bounds__(256)
void gemm_bf16_k(const u16* __restrict__ A, const u16* __restrict__ Bt,
                 const float* __restrict__ bias, void* __restrict__ Out,
                 int M, int N, int K) {
    __shared__ u16 As[2][64 * 32];
    __shared__ u16 Bs[2][128 * 32];
    const int tid  = threadIdx.x;
    const int lane = tid & 31;
    const int wid  = tid >> 5;
    const int wm   = wid >> 1;          // 0..3 : 16-row tile
    const int wn   = wid & 1;           // 0..1 : 64-col strip
    const int half = lane >> 4;
    const int l16  = lane & 15;
    const int mBase = blockIdx.y * 64;
    const int nBase = blockIdx.x * 128;

    const int aRow = tid >> 2;          // 0..63
    const int aCol = (tid & 3) * 8;     // 16B chunk
    const size_t aBase = (size_t)(mBase + aRow) * K + aCol;
    const size_t bBase0 = (size_t)(nBase + aRow) * K + aCol;        // rows 0..63
    const size_t bBase1 = (size_t)(nBase + 64 + aRow) * K + aCol;   // rows 64..127
    const int sA = aRow * 32 + aCol;
    const int sB0 = aRow * 32 + aCol;
    const int sB1 = (64 + aRow) * 32 + aCol;

    const int niter = K / 32;
    u32x4 ra  = *(const u32x4*)&A[aBase];
    u32x4 rb0 = *(const u32x4*)&Bt[bBase0];
    u32x4 rb1 = *(const u32x4*)&Bt[bBase1];

    v8f acc[4] = {{}, {}, {}, {}};
    int cur = 0;
    for (int it = 0; it < niter; ++it) {
        *(u32x4*)&As[cur][sA]  = ra;
        *(u32x4*)&Bs[cur][sB0] = rb0;
        *(u32x4*)&Bs[cur][sB1] = rb1;
        if (it + 1 < niter) {
            const int kc = (it + 1) * 32;
            ra  = *(const u32x4*)&A[aBase + kc];
            rb0 = *(const u32x4*)&Bt[bBase0 + kc];
            rb1 = *(const u32x4*)&Bt[bBase1 + kc];
        }
        __syncthreads();
        const int arow = wm * 16 + l16;
        v8bf a0 = *(const v8bf*)&As[cur][arow * 32 + half * 8];
        v8bf a1 = *(const v8bf*)&As[cur][arow * 32 + 16 + half * 8];
#pragma unroll
        for (int ct = 0; ct < 4; ++ct) {
            const int brow = wn * 64 + ct * 16 + l16;
            v8bf b0 = *(const v8bf*)&Bs[cur][brow * 32 + half * 8];
            v8bf b1 = *(const v8bf*)&Bs[cur][brow * 32 + 16 + half * 8];
            acc[ct] = wmma_bf16(a0, a1, b0, b1, acc[ct]);
        }
        cur ^= 1;
    }
#pragma unroll
    for (int ct = 0; ct < 4; ++ct) {
        for (int r = 0; r < 8; ++r) {
            const int m = mBase + wm * 16 + r + 8 * half;
            const int n = nBase + wn * 64 + ct * 16 + l16;
            float v = acc[ct][r] + (bias ? bias[n] : 0.f);
            if (OUT_BF16) ((u16*)Out)[(size_t)m * N + n] = f2bf(v);
            else          ((float*)Out)[(size_t)m * N + n] = v;
        }
    }
}

// ---------------------------------------------------------------- grid barrier
__device__ __forceinline__ void grid_barrier(u32* bar, u32 nwg) {
    __threadfence();                 // make this thread's stores agent-visible
    __syncthreads();
    if (threadIdx.x == 0) {
        u32 g = __hip_atomic_load(bar + 1, __ATOMIC_RELAXED, __HIP_MEMORY_SCOPE_AGENT);
        u32 arrived = __hip_atomic_fetch_add(bar, 1u, __ATOMIC_ACQ_REL,
                                             __HIP_MEMORY_SCOPE_AGENT);
        if (arrived == nwg - 1) {
            __hip_atomic_store(bar, 0u, __ATOMIC_RELAXED, __HIP_MEMORY_SCOPE_AGENT);
            __hip_atomic_store(bar + 1, g + 1, __ATOMIC_RELEASE,
                               __HIP_MEMORY_SCOPE_AGENT);
        } else {
            while (__hip_atomic_load(bar + 1, __ATOMIC_ACQUIRE,
                                     __HIP_MEMORY_SCOPE_AGENT) == g)
                __builtin_amdgcn_s_sleep(2);
        }
    }
    __syncthreads();
    __threadfence();                 // acquire: drop stale cached h lines
}

// Issue two 16B A-fragment loads (one K-chunk) with explicit asm so the
// scheduler cannot collapse the software pipeline into one register pair.
#define LDH2(lo, hi, O1, O2)                                                   \
    asm volatile("global_load_b128 %0, %2, %3 offset:" O1 "\n\t"               \
                 "global_load_b128 %1, %2, %3 offset:" O2                      \
                 : "=v"(lo), "=v"(hi) : "v"(voff), "s"(hbase))

// Manual loadcnt threshold tied to the registers consumed by the next WMMA.
#define MMA_STEP(i, lo, hi, W)                                                 \
    asm volatile("s_wait_loadcnt " W : "+v"(lo), "+v"(hi));                    \
    acc = wmma_ab(lo, hi, bw0[i], bw1[i], acc)

// ---------------------------------------------------------------- LSTM
// Persistent kernel: 64 WGs x 128 threads.  WG j owns hidden units [8j, 8j+8)
// = a 32-column slice (4 gates x 8 units) of the recurrent weights, held in
// ~128 VGPRs per phase.  A(h) fragments stream through an 8-chunk asm register
// ring with hand-placed s_wait_loadcnt thresholds (~7 chunk-pairs in flight).
__global__ __launch_bounds__(128)
__attribute__((amdgpu_waves_per_eu(1)))
void lstm_persistent_k(const u16* __restrict__ rkT, const u16* __restrict__ wcT,
                       const float* __restrict__ zin, const float* __restrict__ lstm_b,
                       u16* __restrict__ hping, u16* __restrict__ hpong,
                       u16* __restrict__ preds, u32* __restrict__ bar) {
    __shared__ float zbuf[32 * 32];   // [batch][cc]
    __shared__ float cbuf[32 * 8];    // cell state slice [batch][uu]

    const int j    = blockIdx.x;      // 0..63
    const int tid  = threadIdx.x;
    const int lane = tid & 31;
    const int wid  = tid >> 5;        // 0..3
    const int wm   = wid >> 1;        // 0..1 : batch tile
    const int wn   = wid & 1;         // 0..1 : 16-col tile
    const int half = lane >> 4;
    const int l16  = lane & 15;

    const int cc   = wn * 16 + l16;                       // slice column 0..31
    const int gcol = (cc >> 3) * 512 + j * 8 + (cc & 7);  // global gate column
    const int arow = wm * 16 + l16;                       // batch row for A frag
    const u32 voff = (u32)((arow * 512 + half * 8) * 2);  // per-lane byte offset

    // zero h slice and cell state
    for (int c = tid; c < 256; c += 128) {
        int b = c >> 3, uu = c & 7;
        hping[b * 512 + j * 8 + uu] = 0;
        cbuf[c] = 0.f;
    }
    __syncthreads();
    grid_barrier(bar, NWG_LSTM);

    const u16* hin = hping;
    u16* hout = hpong;
    const u16* Wsrc = rkT;
    int stepBeg = 0, stepEnd = HH;

    for (int phase = 0; phase < 2; ++phase) {
        // hoist this wave's B fragments (16 K-chunks) into registers
        v8bf bw0[16], bw1[16];
#pragma unroll
        for (int i = 0; i < 16; ++i) {
            const u16* wp = &Wsrc[(size_t)gcol * 512 + i * 32 + half * 8];
            bw0[i] = *(const v8bf*)wp;
            bw1[i] = *(const v8bf*)(wp + 16);
        }
        for (int step = stepBeg; step < stepEnd; ++step) {
            // z[32][32] = h[32][512] @ Wslice[512][32]
            const u64 hbase = (u64)(uintptr_t)hin;
            u32x4 l0, h0, l1, h1, l2, h2, l3, h3, l4, h4, l5, h5, l6, h6, l7, h7;
            // prologue: chunks 0..7 (16 loads in flight)
            LDH2(l0, h0,   "0",  "32");
            LDH2(l1, h1,  "64",  "96");
            LDH2(l2, h2, "128", "160");
            LDH2(l3, h3, "192", "224");
            LDH2(l4, h4, "256", "288");
            LDH2(l5, h5, "320", "352");
            LDH2(l6, h6, "384", "416");
            LDH2(l7, h7, "448", "480");
            v8f acc = {};
            // steady state: consume chunk i, refill chunk i+8 into same slot
            MMA_STEP(0, l0, h0, "14");  LDH2(l0, h0, "512", "544");
            MMA_STEP(1, l1, h1, "14");  LDH2(l1, h1, "576", "608");
            MMA_STEP(2, l2, h2, "14");  LDH2(l2, h2, "640", "672");
            MMA_STEP(3, l3, h3, "14");  LDH2(l3, h3, "704", "736");
            MMA_STEP(4, l4, h4, "14");  LDH2(l4, h4, "768", "800");
            MMA_STEP(5, l5, h5, "14");  LDH2(l5, h5, "832", "864");
            MMA_STEP(6, l6, h6, "14");  LDH2(l6, h6, "896", "928");
            MMA_STEP(7, l7, h7, "14");  LDH2(l7, h7, "960", "992");
            // drain
            MMA_STEP(8,  l0, h0, "14");
            MMA_STEP(9,  l1, h1, "12");
            MMA_STEP(10, l2, h2, "10");
            MMA_STEP(11, l3, h3, "8");
            MMA_STEP(12, l4, h4, "6");
            MMA_STEP(13, l5, h5, "4");
            MMA_STEP(14, l6, h6, "2");
            MMA_STEP(15, l7, h7, "0");
#pragma unroll
            for (int r = 0; r < 8; ++r) {
                const int b = wm * 16 + r + 8 * half;
                float z = acc[r];
                if (phase == 0) z += zin[(size_t)(b * HH + step) * GG + gcol];
                else            z += lstm_b[gcol];
                zbuf[b * 32 + cc] = z;
            }
            __syncthreads();
            // gates: 32 batches x 8 units
            for (int c = tid; c < 256; c += 128) {
                int b = c >> 3, uu = c & 7;
                float zi = zbuf[b * 32 + uu];
                float zf = zbuf[b * 32 + 8 + uu];
                float zg = zbuf[b * 32 + 16 + uu];
                float zo = zbuf[b * 32 + 24 + uu];
                float ig = 1.f / (1.f + __expf(-zi));
                float fg = 1.f / (1.f + __expf(-zf));
                float gg = tanhf(zg);
                float og = 1.f / (1.f + __expf(-zo));
                float cn = fg * cbuf[c] + ig * gg;
                float hn = og * tanhf(cn);
                cbuf[c] = cn;
                u16 hb = f2bf(hn);
                hout[b * 512 + j * 8 + uu] = hb;
                if (phase == 1)
                    preds[(size_t)(b * PP + (step - HH)) * 512 + j * 8 + uu] = hb;
            }
            grid_barrier(bar, NWG_LSTM);
            const u16* t = hin; hin = hout; hout = (u16*)t;
        }
        Wsrc = wcT; stepBeg = HH; stepEnd = HH + PP;
    }
}

// ---------------------------------------------------------------- launch
extern "C" void kernel_launch(void* const* d_in, const int* in_sizes, int n_in,
                              void* d_out, int out_size, void* d_ws, size_t ws_size,
                              hipStream_t stream) {
    (void)in_sizes; (void)n_in; (void)out_size; (void)ws_size;
    const float* x       = (const float*)d_in[0];   // [32,168,512,8]
    const float* conv_w  = (const float*)d_in[1];   // [4096,512]
    const float* conv_b  = (const float*)d_in[2];   // [512]
    const float* lstm_k  = (const float*)d_in[3];   // [512,2048]
    const float* lstm_rk = (const float*)d_in[4];   // [512,2048]
    const float* lstm_b  = (const float*)d_in[5];   // [2048]
    const float* dense_w = (const float*)d_in[6];   // [512,512]
    const float* dense_b = (const float*)d_in[7];   // [512]
    float* out = (float*)d_out;                     // [32,24,512]

    char* w = (char*)d_ws; size_t off = 0;
    auto alloc = [&](size_t bytes) { void* p = w + off; off = (off + bytes + 255) & ~(size_t)255; return p; };
    u16* x_bf  = (u16*)alloc((size_t)MROWS * NF * 2);
    u16* convT = (u16*)alloc((size_t)NN * NF * 2);
    u16* kT    = (u16*)alloc((size_t)GG * NN * 2);
    u16* rkT   = (u16*)alloc((size_t)GG * NN * 2);
    u16* wcT   = (u16*)alloc((size_t)GG * NN * 2);
    u16* dT    = (u16*)alloc((size_t)NN * NN * 2);
    u16* xr    = (u16*)alloc((size_t)MROWS * NN * 2);
    float* zin = (float*)alloc((size_t)MROWS * GG * 4);
    u16* hping = (u16*)alloc((size_t)BB * NN * 2);
    u16* hpong = (u16*)alloc((size_t)BB * NN * 2);
    u16* preds = (u16*)alloc((size_t)BB * PP * NN * 2);
    u32* bar   = (u32*)alloc(256);

    init_bar_k<<<1, 64, 0, stream>>>(bar);

    const int nX = MROWS * NF;
    cvt_bf16_k<<<4096, 256, 0, stream>>>(x, x_bf, nX);
    transpose_bf16_k<<<(NF * NN + 255) / 256, 256, 0, stream>>>(conv_w, convT, NF, NN);
    transpose_bf16_k<<<(NN * GG + 255) / 256, 256, 0, stream>>>(lstm_k, kT, NN, GG);
    transpose_bf16_k<<<(NN * GG + 255) / 256, 256, 0, stream>>>(lstm_rk, rkT, NN, GG);
    transpose_sum_bf16_k<<<(NN * GG + 255) / 256, 256, 0, stream>>>(lstm_k, lstm_rk, wcT, NN, GG);
    transpose_bf16_k<<<(NN * NN + 255) / 256, 256, 0, stream>>>(dense_w, dT, NN, NN);

    // xr[5376][512] = x_bf @ conv_w + conv_b   (bf16 out)
    gemm_bf16_k<1><<<dim3(NN / 128, MROWS / 64), 256, 0, stream>>>(
        x_bf, convT, conv_b, xr, MROWS, NN, NF);
    // zin[5376][2048] = xr @ lstm_k + lstm_b   (f32 out)
    gemm_bf16_k<0><<<dim3(GG / 128, MROWS / 64), 256, 0, stream>>>(
        xr, kT, lstm_b, zin, MROWS, GG, NN);
    // sequential encoder (168) + decoder (24) in one persistent kernel
    lstm_persistent_k<<<NWG_LSTM, 128, 0, stream>>>(
        rkT, wcT, zin, lstm_b, hping, hpong, preds, bar);
    // out[768][512] = preds @ dense_w + dense_b
    gemm_bf16_k<0><<<dim3(NN / 128, (BB * PP) / 64), 256, 0, stream>>>(
        preds, dT, dense_b, out, BB * PP, NN, NN);
}
